// EfficientAttention_76424648065618
// MI455X (gfx1250) — matrix operand order repaired and checked by reading, not compiled, not run on hardware
//
#include <hip/hip_runtime.h>
#include <hip/hip_bf16.h>

// ---------------- problem constants ----------------
#define B_  2
#define N_  2048
#define C_  1024
#define H_  16
#define DH  64
#define C3  3072   // 3*C_

// workspace layout (bytes). Total 48 MB.
#define OFF_XBF   ((size_t)0)                       // x  bf16  [B,N,C]      8 MB
#define OFF_WQKV  ((size_t)8388608)                 // w_qkv bf16 [3C,C]     6 MB
#define OFF_WPRJ  ((size_t)14680064)                // w_proj bf16 [C,C]     2 MB
#define OFF_Q     ((size_t)16777216)                // Q bf16 [B,H,N,Dh]    8 MB (pre-scaled by 1/8)
#define OFF_K     ((size_t)25165824)                // K bf16 [B,H,N,Dh]    8 MB
#define OFF_VT    ((size_t)33554432)                // V^T bf16 [B,H,Dh,N]  8 MB
#define OFF_AO    ((size_t)41943040)                // attn out bf16 [B,N,C] 8 MB

typedef __attribute__((ext_vector_type(16))) __bf16 v16bf;
typedef __attribute__((ext_vector_type(8)))  __bf16 v8bf;
typedef __attribute__((ext_vector_type(8)))  float  v8f;
typedef __attribute__((ext_vector_type(4)))  unsigned int u32x4;
typedef __attribute__((ext_vector_type(8)))  int i32x8;
typedef __attribute__((ext_vector_type(4)))  int i32x4;

// Build a 16-elem bf16 A/B fragment from two contiguous 16-byte chunks
// (works for both global and LDS pointers).
__device__ __forceinline__ v16bf load_frag(const __hip_bfloat16* plo,
                                           const __hip_bfloat16* phi) {
  v8bf lo = *(const v8bf*)plo;
  v8bf hi = *(const v8bf*)phi;
  v16bf r;
#pragma unroll
  for (int i = 0; i < 8; ++i) { r[i] = lo[i]; r[i + 8] = hi[i]; }
  return r;
}

__device__ __forceinline__ v8f wmma_bf16(v16bf a, v16bf b, v8f c) {
  return __builtin_amdgcn_wmma_f32_16x16x32_bf16(false, a, false, b,
                                                 (short)0, c, false, false);
}

// ---------------- Tensor Data Mover: 2D bf16 tile -> LDS ----------------
// D# per CDNA5 ISA. Tensor dims == tile dims
// (tile fits exactly; no OOB path). stride0 = elements between rows.
// tile_d0 = contiguous elements per row, tile_d1 = rows.
__device__ __forceinline__ void tdm_load_tile_2d(__hip_bfloat16* lds_dst,
                                                 const __hip_bfloat16* gsrc,
                                                 unsigned tile_d0,
                                                 unsigned tile_d1,
                                                 unsigned long long stride0) {
  unsigned long long ga = (unsigned long long)(size_t)gsrc;
  unsigned lds_off = (unsigned)(size_t)lds_dst;   // low 32 bits = LDS offset
  u32x4 g0;
  g0[0] = 1u;                                     // count=1, user descriptor
  g0[1] = lds_off;                                // lds_addr
  g0[2] = (unsigned)(ga & 0xffffffffu);           // global_addr[31:0]
  g0[3] = (unsigned)((ga >> 32) & 0x01ffffffu) | (2u << 30);  // addr hi | type=2
  i32x8 g1;
  g1[0] = (int)(1u << 16);                        // data_size=1 -> 2 bytes
  g1[1] = (int)((tile_d0 & 0xffffu) << 16);       // tensor_dim0[15:0]
  g1[2] = (int)(((tile_d0 >> 16) & 0xffffu) | ((tile_d1 & 0xffffu) << 16));
  g1[3] = (int)(((tile_d1 >> 16) & 0xffffu) | ((tile_d0 & 0xffffu) << 16)); // tile_dim0
  g1[4] = (int)(tile_d1 & 0xffffu);               // tile_dim1 (tile_dim2=0)
  g1[5] = (int)(unsigned)(stride0 & 0xffffffffu); // tensor_dim0_stride[31:0]
  g1[6] = (int)((stride0 >> 32) & 0xffffu);       // stride[47:32]; dim1_stride=0
  g1[7] = 0;
  i32x4 z4 = {0, 0, 0, 0};
#if __clang_major__ >= 23
  i32x8 z8 = {0, 0, 0, 0, 0, 0, 0, 0};
  __builtin_amdgcn_tensor_load_to_lds(g0, g1, z4, z4, z8, 0);
#else
  __builtin_amdgcn_tensor_load_to_lds(g0, g1, z4, z4, 0);
#endif
}

// ---------------- fp32 -> bf16 convert ----------------
__global__ void attn_cvt_bf16(const float* __restrict__ src,
                              __hip_bfloat16* __restrict__ dst, int n) {
  int i = (blockIdx.x * blockDim.x + threadIdx.x) * 4;
  if (i < n) {
    float4 v = *(const float4*)(src + i);
    dst[i + 0] = __float2bfloat16(v.x);
    dst[i + 1] = __float2bfloat16(v.y);
    dst[i + 2] = __float2bfloat16(v.z);
    dst[i + 3] = __float2bfloat16(v.w);
  }
}

// ---------------- QKV projection GEMM ----------------
// grid = (4096/128, 3072/64), block = 128 (4 waves; wave does 32 rows x 64 cols)
// Weight tile (64 cols x 32 k) staged via TDM into LDS, double-buffered.
__global__ __launch_bounds__(128) void attn_qkv_gemm(
    const __hip_bfloat16* __restrict__ xbf,
    const __hip_bfloat16* __restrict__ wbf,
    const float* __restrict__ bqkv,
    __hip_bfloat16* __restrict__ Q,
    __hip_bfloat16* __restrict__ K,
    __hip_bfloat16* __restrict__ Vt) {
  __shared__ __align__(16) __hip_bfloat16 wsh[2][64 * 32];

  const int wave = threadIdx.x >> 5, lane = threadIdx.x & 31;
  const int nl = lane & 15, half = lane >> 4;
  const int rbase = blockIdx.x * 128 + wave * 32;
  const int cbase = blockIdx.y * 64;

  v8f acc0[4] = {}, acc1[4] = {};
  const __hip_bfloat16* arow0 = xbf + (size_t)(rbase + nl) * C_;
  const __hip_bfloat16* arow1 = xbf + (size_t)(rbase + 16 + nl) * C_;
  const __hip_bfloat16* wtile = wbf + (size_t)cbase * C_;

  if (wave == 0) tdm_load_tile_2d(wsh[0], wtile, 32, 64, C_);

  for (int k0 = 0; k0 < C_; k0 += 32) {
    const int buf = (k0 >> 5) & 1;
    if (wave == 0) {
      if (k0 + 32 < C_) {
        tdm_load_tile_2d(wsh[buf ^ 1], wtile + (k0 + 32), 32, 64, C_);
        __builtin_amdgcn_s_wait_tensorcnt(1);
      } else {
        __builtin_amdgcn_s_wait_tensorcnt(0);
      }
    }
    __syncthreads();

    v16bf a0 = load_frag(arow0 + k0 + 16 * half, arow0 + k0 + 16 * half + 8);
    v16bf a1 = load_frag(arow1 + k0 + 16 * half, arow1 + k0 + 16 * half + 8);
#pragma unroll
    for (int cc = 0; cc < 4; ++cc) {
      const __hip_bfloat16* brow = &wsh[buf][(cc * 16 + nl) * 32 + 16 * half];
      v16bf b = load_frag(brow, brow + 8);
      acc0[cc] = wmma_bf16(a0, b, acc0[cc]);
      acc1[cc] = wmma_bf16(a1, b, acc1[cc]);
    }
    __syncthreads();
  }

  const int s = cbase / C_;  // 0=Q 1=K 2=V  (uniform: 64 | 1024)
#pragma unroll
  for (int cc = 0; cc < 4; ++cc) {
    const int col = cbase + cc * 16 + nl;
    const float bias = bqkv[col];
    const int colm = col - s * C_;
    const int h = colm >> 6, d = colm & 63;
#pragma unroll
    for (int g = 0; g < 2; ++g) {
#pragma unroll
      for (int r = 0; r < 8; ++r) {
        const int row = rbase + g * 16 + half * 8 + r;
        const int bb = row >> 11;        // row / N_
        const int tt = row & (N_ - 1);   // row % N_
        const float v = (g ? acc1[cc][r] : acc0[cc][r]) + bias;
        if (s == 0)
          Q[(((size_t)bb * H_ + h) * N_ + tt) * DH + d] =
              __float2bfloat16(v * 0.125f);  // fold Dh^-0.5
        else if (s == 1)
          K[(((size_t)bb * H_ + h) * N_ + tt) * DH + d] = __float2bfloat16(v);
        else
          Vt[(((size_t)bb * H_ + h) * DH + d) * N_ + tt] = __float2bfloat16(v);
      }
    }
  }
}

// ---------------- flash attention core ----------------
// grid = B*H*(N/64) = 1024 blocks, block = 128 (4 waves; wave owns 16 query rows)
// K tile (32 keys x 64) and V^T tile (64 x 32 keys) staged via TDM, double-buffered.
__global__ __launch_bounds__(128) void attn_flash(
    const __hip_bfloat16* __restrict__ Q,
    const __hip_bfloat16* __restrict__ K,
    const __hip_bfloat16* __restrict__ Vt,
    __hip_bfloat16* __restrict__ attn_out) {
  __shared__ __align__(16) __hip_bfloat16 kbuf[2][32 * 64];
  __shared__ __align__(16) __hip_bfloat16 vbuf[2][64 * 32];
  __shared__ __align__(16) __hip_bfloat16 plds[4][16 * 32];

  const int wave = threadIdx.x >> 5, lane = threadIdx.x & 31;
  const int nl = lane & 15, half = lane >> 4;
  const int bh = blockIdx.x >> 5;     // b*H + h
  const int qblk = blockIdx.x & 31;
  const int qbase = qblk * 64 + wave * 16;

  const __hip_bfloat16* Qh = Q + (size_t)bh * N_ * DH;
  const __hip_bfloat16* Kh = K + (size_t)bh * N_ * DH;
  const __hip_bfloat16* Vh = Vt + (size_t)bh * DH * N_;

  // Q fragments (16 rows x Dh=64, two K=32 chunks) — loaded once from global.
  const __hip_bfloat16* qrow = Qh + (size_t)(qbase + nl) * DH;
  const v16bf aq0 = load_frag(qrow + 16 * half, qrow + 8 + 16 * half);
  const v16bf aq1 = load_frag(qrow + 32 + 16 * half, qrow + 40 + 16 * half);

  float mrun[8], lrun[8];
#pragma unroll
  for (int r = 0; r < 8; ++r) { mrun[r] = -1.0e30f; lrun[r] = 0.0f; }
  v8f oacc[4] = {};

  __hip_bfloat16* pw = plds[wave];

  if (wave == 0) {
    tdm_load_tile_2d(kbuf[0], Kh, 64, 32, DH);   // 32 key rows of 64
    tdm_load_tile_2d(vbuf[0], Vh, 32, 64, N_);   // 64 d rows of 32 keys
  }

  for (int kb = 0; kb < N_; kb += 32) {
    const int buf = (kb >> 5) & 1;
    if (wave == 0) {
      if (kb + 32 < N_) {
        tdm_load_tile_2d(kbuf[buf ^ 1], Kh + (size_t)(kb + 32) * DH, 64, 32, DH);
        tdm_load_tile_2d(vbuf[buf ^ 1], Vh + (kb + 32), 32, 64, N_);
        __builtin_amdgcn_s_wait_tensorcnt(2);  // current pair done; next in flight
      } else {
        __builtin_amdgcn_s_wait_tensorcnt(0);
      }
    }
    __syncthreads();

    // ---- S = (Q*scale) @ K^T for 32 keys: two 16x16 C tiles ----
    const __hip_bfloat16* kr0 = &kbuf[buf][nl * 64 + 16 * half];
    const __hip_bfloat16* kr1 = &kbuf[buf][(16 + nl) * 64 + 16 * half];
    v16bf b00 = load_frag(kr0, kr0 + 8);        // d 0..31
    v16bf b01 = load_frag(kr0 + 32, kr0 + 40);  // d 32..63
    v16bf b10 = load_frag(kr1, kr1 + 8);
    v16bf b11 = load_frag(kr1 + 32, kr1 + 40);
    v8f s0 = {}, s1 = {};
    s0 = wmma_bf16(aq0, b00, s0);
    s0 = wmma_bf16(aq1, b01, s0);
    s1 = wmma_bf16(aq0, b10, s1);
    s1 = wmma_bf16(aq1, b11, s1);

    // ---- online softmax over the 32-key slab ----
#pragma unroll
    for (int r = 0; r < 8; ++r) {
      float mt = fmaxf(s0[r], s1[r]);
      mt = fmaxf(mt, __shfl_xor(mt, 1));
      mt = fmaxf(mt, __shfl_xor(mt, 2));
      mt = fmaxf(mt, __shfl_xor(mt, 4));
      mt = fmaxf(mt, __shfl_xor(mt, 8));
      const float mnew = fmaxf(mrun[r], mt);
      const float sc = __expf(mrun[r] - mnew);
      const float p0 = __expf(s0[r] - mnew);
      const float p1 = __expf(s1[r] - mnew);
      float rs = p0 + p1;
      rs += __shfl_xor(rs, 1);
      rs += __shfl_xor(rs, 2);
      rs += __shfl_xor(rs, 4);
      rs += __shfl_xor(rs, 8);
      lrun[r] = lrun[r] * sc + rs;
      mrun[r] = mnew;
      oacc[0][r] *= sc; oacc[1][r] *= sc;
      oacc[2][r] *= sc; oacc[3][r] *= sc;
      // stash P row-major [16][32] in wave-private LDS (C->A layout transpose;
      // same-wave DS ops are in-order, no barrier needed)
      const int row = half * 8 + r;
      pw[row * 32 + nl] = __float2bfloat16(p0);
      pw[row * 32 + 16 + nl] = __float2bfloat16(p1);
    }

    // ---- O += P @ V  (A = P 16x32, B = V^T slabs 32x16 per d-chunk) ----
    const __hip_bfloat16* prow = pw + nl * 32;
    v16bf pa = load_frag(prow + 16 * half, prow + 8 + 16 * half);
#pragma unroll
    for (int cc = 0; cc < 4; ++cc) {
      const __hip_bfloat16* vr = &vbuf[buf][(cc * 16 + nl) * 32 + 16 * half];
      v16bf bv = load_frag(vr, vr + 8);
      oacc[cc] = wmma_bf16(pa, bv, oacc[cc]);
    }
    __syncthreads();
  }

  // ---- normalize and write bf16 [B,N,C] for the projection GEMM ----
  const int bb = bh >> 4;   // / H_
  const int h = bh & 15;
#pragma unroll
  for (int cc = 0; cc < 4; ++cc) {
    const int col = h * 64 + cc * 16 + nl;
#pragma unroll
    for (int r = 0; r < 8; ++r) {
      const int tok = qbase + half * 8 + r;
      attn_out[((size_t)bb * N_ + tok) * C_ + col] =
          __float2bfloat16(oacc[cc][r] / lrun[r]);
    }
  }
}

// ---------------- output projection GEMM ----------------
// out f32 [4096,1024] = attn_out_bf16 @ w_proj^T + b_proj
// grid = (32,16), block = 128; weight tile via TDM, double-buffered.
__global__ __launch_bounds__(128) void attn_proj_gemm(
    const __hip_bfloat16* __restrict__ abf,
    const __hip_bfloat16* __restrict__ wbf,
    const float* __restrict__ bproj,
    float* __restrict__ out) {
  __shared__ __align__(16) __hip_bfloat16 wsh[2][64 * 32];

  const int wave = threadIdx.x >> 5, lane = threadIdx.x & 31;
  const int nl = lane & 15, half = lane >> 4;
  const int rbase = blockIdx.x * 128 + wave * 32;
  const int cbase = blockIdx.y * 64;

  v8f acc0[4] = {}, acc1[4] = {};
  const __hip_bfloat16* arow0 = abf + (size_t)(rbase + nl) * C_;
  const __hip_bfloat16* arow1 = abf + (size_t)(rbase + 16 + nl) * C_;
  const __hip_bfloat16* wtile = wbf + (size_t)cbase * C_;

  if (wave == 0) tdm_load_tile_2d(wsh[0], wtile, 32, 64, C_);

  for (int k0 = 0; k0 < C_; k0 += 32) {
    const int buf = (k0 >> 5) & 1;
    if (wave == 0) {
      if (k0 + 32 < C_) {
        tdm_load_tile_2d(wsh[buf ^ 1], wtile + (k0 + 32), 32, 64, C_);
        __builtin_amdgcn_s_wait_tensorcnt(1);
      } else {
        __builtin_amdgcn_s_wait_tensorcnt(0);
      }
    }
    __syncthreads();

    v16bf a0 = load_frag(arow0 + k0 + 16 * half, arow0 + k0 + 16 * half + 8);
    v16bf a1 = load_frag(arow1 + k0 + 16 * half, arow1 + k0 + 16 * half + 8);
#pragma unroll
    for (int cc = 0; cc < 4; ++cc) {
      const __hip_bfloat16* brow = &wsh[buf][(cc * 16 + nl) * 32 + 16 * half];
      v16bf b = load_frag(brow, brow + 8);
      acc0[cc] = wmma_bf16(a0, b, acc0[cc]);
      acc1[cc] = wmma_bf16(a1, b, acc1[cc]);
    }
    __syncthreads();
  }

#pragma unroll
  for (int cc = 0; cc < 4; ++cc) {
    const int col = cbase + cc * 16 + nl;
    const float bias = bproj[col];
#pragma unroll
    for (int r = 0; r < 8; ++r) {
      const int row0 = rbase + half * 8 + r;
      out[(size_t)row0 * C_ + col] = acc0[cc][r] + bias;
      const int row1 = rbase + 16 + half * 8 + r;
      out[(size_t)row1 * C_ + col] = acc1[cc][r] + bias;
    }
  }
}

// ---------------- launcher ----------------
extern "C" void kernel_launch(void* const* d_in, const int* in_sizes, int n_in,
                              void* d_out, int out_size, void* d_ws,
                              size_t ws_size, hipStream_t stream) {
  (void)in_sizes; (void)n_in; (void)out_size; (void)ws_size;
  const float* x      = (const float*)d_in[0];
  const float* w_qkv  = (const float*)d_in[1];
  const float* b_qkv  = (const float*)d_in[2];
  const float* w_proj = (const float*)d_in[3];
  const float* b_proj = (const float*)d_in[4];
  float* out = (float*)d_out;

  char* ws = (char*)d_ws;
  __hip_bfloat16* xbf = (__hip_bfloat16*)(ws + OFF_XBF);
  __hip_bfloat16* wqk = (__hip_bfloat16*)(ws + OFF_WQKV);
  __hip_bfloat16* wpr = (__hip_bfloat16*)(ws + OFF_WPRJ);
  __hip_bfloat16* Q   = (__hip_bfloat16*)(ws + OFF_Q);
  __hip_bfloat16* K   = (__hip_bfloat16*)(ws + OFF_K);
  __hip_bfloat16* Vt  = (__hip_bfloat16*)(ws + OFF_VT);
  __hip_bfloat16* ao  = (__hip_bfloat16*)(ws + OFF_AO);

  const int nx = B_ * N_ * C_;
  const int nw1 = C3 * C_;
  const int nw2 = C_ * C_;
  attn_cvt_bf16<<<(nx / 4 + 255) / 256, 256, 0, stream>>>(x, xbf, nx);
  attn_cvt_bf16<<<(nw1 / 4 + 255) / 256, 256, 0, stream>>>(w_qkv, wqk, nw1);
  attn_cvt_bf16<<<(nw2 / 4 + 255) / 256, 256, 0, stream>>>(w_proj, wpr, nw2);

  attn_qkv_gemm<<<dim3((B_ * N_) / 128, C3 / 64), 128, 0, stream>>>(
      xbf, wqk, b_qkv, Q, K, Vt);

  attn_flash<<<B_ * H_ * (N_ / 64), 128, 0, stream>>>(Q, K, Vt, ao);

  attn_proj_gemm<<<dim3((B_ * N_) / 128, C_ / 64), 128, 0, stream>>>(
      ao, wpr, b_proj, out);
}
